// MultiHeadLatentAttention_56014963475108
// MI455X (gfx1250) — matrix operand order repaired and checked by reading, not compiled, not run on hardware
//
#include <hip/hip_runtime.h>
#include <hip/hip_bf16.h>

// ---------------------------------------------------------------------------
// MLA forward for MI455X (gfx1250, wave32, WMMA).  All heavy math runs through
// v_wmma_f32_16x16x32_bf16 with fp32 accumulation; intermediates are bf16.
// Pure-copy tile staging uses GLOBAL_LOAD_ASYNC_TO_LDS (ASYNCcnt) when the
// toolchain exposes the builtins.
// ---------------------------------------------------------------------------

#define BATCH     2
#define SEQ       2048
#define DMODEL    2048
#define NHEADS    16
#define DHEAD     128
#define DLAT      512
#define DLATQ     1536
#define DROPE     64
#define DQK       192          // DHEAD + DROPE
#define QROW      (NHEADS * DQK)   // 3072
#define VROW      (NHEADS * DHEAD) // 2048
#define MTOT      (BATCH * SEQ)    // 4096

typedef __attribute__((ext_vector_type(16))) __bf16 bf16x16;
typedef __attribute__((ext_vector_type(8)))  float  floatx8;
typedef __attribute__((ext_vector_type(4)))  int    i32x4;

union AFrag { bf16x16 v; unsigned int u[8]; };

#if defined(__has_builtin)
#if __has_builtin(__builtin_amdgcn_global_load_async_to_lds_b128) && \
    __has_builtin(__builtin_amdgcn_s_wait_asynccnt)
#define USE_ASYNC_LDS 1
#endif
#endif
#ifndef USE_ASYNC_LDS
#define USE_ASYNC_LDS 0
#endif

#if USE_ASYNC_LDS
// 16-byte global -> LDS direct copy (no VGPR data), tracked by ASYNCcnt.
// Builtin signature (probe-confirmed by diagnostic): (v4i AS1*, v4i AS3*, Ii, Ii)
typedef __attribute__((address_space(1))) i32x4 gi32x4;
typedef __attribute__((address_space(3))) i32x4 li32x4;
__device__ __forceinline__ void async_copy16(const unsigned short* g,
                                             unsigned short* l) {
    __builtin_amdgcn_global_load_async_to_lds_b128(
        (gi32x4*)g, (li32x4*)l, 0, 0);
}
#endif

__device__ __forceinline__ unsigned short f32_to_bf16(float f) {
    union { float f; unsigned int u; } x; x.f = f;
    unsigned int r = x.u + 0x7FFFu + ((x.u >> 16) & 1u);
    return (unsigned short)(r >> 16);
}
__device__ __forceinline__ float bf16_to_f32(unsigned short h) {
    union { unsigned int u; float f; } x; x.u = ((unsigned int)h) << 16;
    return x.f;
}

// ------------------------------ fp32 -> bf16 -------------------------------
__global__ void cvt_f32_bf16_kernel(const float* __restrict__ src,
                                    unsigned short* __restrict__ dst, int n) {
    for (int i = blockIdx.x * blockDim.x + threadIdx.x; i < n;
         i += gridDim.x * blockDim.x)
        dst[i] = f32_to_bf16(src[i]);
}

// ------------------------------ WMMA GEMM ----------------------------------
// C[M,N] = A[M,K] * B[K,N]; A,B bf16 row-major.  Output either bf16 with a
// column remap (col -> (col/grp)*gstr + goff + col%grp, row stride ldc) or
// plain fp32.  Block = 256 threads = 8 waves; tile 128x128x32; each wave owns
// a 64x32 C sub-tile (4x2 WMMA accumulators).
template <bool OUT_F32>
__global__ void __launch_bounds__(256)
gemm_bf16_wmma(const unsigned short* __restrict__ A,
               const unsigned short* __restrict__ Bm,
               void* __restrict__ C,
               int M, int N, int K,
               int ldc, int grp, int gstr, int goff)
{
    constexpr int BM = 128, BN = 128, BK = 32;
    __shared__ unsigned short As[BM][BK + 8];   // [m][k], 80B row stride (16B ok)
    __shared__ unsigned short Bs[BN][BK + 2];   // transposed: [n][k]

    const int tid  = threadIdx.x;
    const int wave = tid >> 5, lane = tid & 31;
    const int waveM = wave >> 2;                // 0..1
    const int waveN = wave & 3;                 // 0..3
    const int bm0 = blockIdx.x * BM;
    const int bn0 = blockIdx.y * BN;

    floatx8 zero = {};
    floatx8 acc[4][2];
    #pragma unroll
    for (int i = 0; i < 4; ++i)
        #pragma unroll
        for (int j = 0; j < 2; ++j) acc[i][j] = zero;

    const int mrow  = lane & 15;
    const int halfK = (lane >> 4) * 8;

    for (int kb = 0; kb < K; kb += BK) {
        // ---- stage A tile: thread -> (row, half-row of 16 bf16) ----
        {
            int m  = tid >> 1;
            int kh = (tid & 1) * 16;
            const unsigned short* g = &A[(size_t)(bm0 + m) * K + kb + kh];
#if USE_ASYNC_LDS
            async_copy16(g,     &As[m][kh]);       // global -> LDS, no VGPRs
            async_copy16(g + 8, &As[m][kh + 8]);
#else
            const uint4* gp = reinterpret_cast<const uint4*>(g);
            uint4 d0 = gp[0], d1 = gp[1];
            unsigned int* ls = reinterpret_cast<unsigned int*>(&As[m][kh]);
            ls[0] = d0.x; ls[1] = d0.y; ls[2] = d0.z; ls[3] = d0.w;
            ls[4] = d1.x; ls[5] = d1.y; ls[6] = d1.z; ls[7] = d1.w;
#endif
        }
        // ---- stage B tile transposed (guard partial N, e.g. W_kr N=64) ----
        {
            int k  = tid >> 3;
            int n0 = (tid & 7) * 16;
            #pragma unroll
            for (int e = 0; e < 16; ++e) {
                int n  = n0 + e;
                int gn = bn0 + n;
                Bs[n][k] = (gn < N) ? Bm[(size_t)(kb + k) * N + gn]
                                    : (unsigned short)0;
            }
        }
        if (kb + BK < K) {  // global_prefetch_b8 of next slab
            __builtin_prefetch(&A[(size_t)(bm0 + (tid >> 1)) * K + kb + BK], 0, 1);
            __builtin_prefetch(&Bm[(size_t)(kb + BK + (tid >> 3)) * N + bn0], 0, 1);
        }
#if USE_ASYNC_LDS
        __builtin_amdgcn_s_wait_asynccnt(0);
#endif
        __syncthreads();

        // ---- A fragments (16x32, documented 16-bit A layout) ----
        AFrag a[4];
        #pragma unroll
        for (int i = 0; i < 4; ++i) {
            const unsigned short* ap = &As[waveM * 64 + i * 16 + mrow][0];
            #pragma unroll
            for (int vv = 0; vv < 4; ++vv)
                a[i].u[vv] = *reinterpret_cast<const unsigned int*>(&ap[halfK + 2 * vv]);
            #pragma unroll
            for (int vv = 0; vv < 4; ++vv)
                a[i].u[4 + vv] =
                    *reinterpret_cast<const unsigned int*>(&ap[16 + halfK + 2 * vv]);
        }
        // ---- B fragments (32x16) ----
        AFrag b[2];
        const int bHalf = (lane >> 4) * 8;
        #pragma unroll
        for (int j = 0; j < 2; ++j) {
            const unsigned short* bp = &Bs[waveN * 32 + j * 16 + (lane & 15)][0];
            #pragma unroll
            for (int vv = 0; vv < 8; ++vv)
                b[j].u[vv] =
                    *reinterpret_cast<const unsigned int*>(&bp[bHalf + 2 * vv + (vv >= 4 ? 8 : 0)]);
        }
        #pragma unroll
        for (int i = 0; i < 4; ++i)
            #pragma unroll
            for (int j = 0; j < 2; ++j)
                acc[i][j] = __builtin_amdgcn_wmma_f32_16x16x32_bf16(
                    false, a[i].v, false, b[j].v, (short)0, acc[i][j], false, false);
        __syncthreads();
    }

    // ---- epilogue: C layout lane(n)=lane%16, row = 8*(lane/16)+r ----
    const int rgrp = (lane >> 4) * 8;
    const int ncol = lane & 15;
    #pragma unroll
    for (int i = 0; i < 4; ++i)
        #pragma unroll
        for (int j = 0; j < 2; ++j)
            #pragma unroll
            for (int r = 0; r < 8; ++r) {
                int row = bm0 + waveM * 64 + i * 16 + rgrp + r;
                int col = bn0 + waveN * 32 + j * 16 + ncol;
                if (col < N && row < M) {
                    float vv = acc[i][j][r];
                    size_t idx = (size_t)row * ldc +
                                 (size_t)(col / grp) * gstr + goff + (col % grp);
                    if constexpr (OUT_F32)
                        ((float*)C)[idx] = vv;
                    else
                        ((unsigned short*)C)[idx] = f32_to_bf16(vv);
                }
            }
}

// ------------------------------ RoPE ---------------------------------------
// In-place rotate q[..., h, 128:192]; rotate kr_tmp[bt, 64] and broadcast to
// every head of k[..., h, 128:192].
__global__ void rope_kernel(unsigned short* __restrict__ q,
                            unsigned short* __restrict__ k,
                            const unsigned short* __restrict__ kr)
{
    const int bt  = blockIdx.x;            // 0..MTOT-1
    const int t   = bt % SEQ;
    const int tid = threadIdx.x;
    const float LOG_BASE = 9.2103403719761836f; // ln(10000)

    for (int p = tid; p < NHEADS * 32; p += blockDim.x) {
        int h = p >> 5, i = p & 31;
        float fr  = __expf(-((float)(2 * i) / 64.f) * LOG_BASE);
        float ang = (float)t * fr;
        float c = __cosf(ang), s = __sinf(ang);
        unsigned short* base = &q[(size_t)bt * QROW + h * DQK + DHEAD + 2 * i];
        float x1 = bf16_to_f32(base[0]), x2 = bf16_to_f32(base[1]);
        base[0] = f32_to_bf16(x1 * c - x2 * s);
        base[1] = f32_to_bf16(x1 * s + x2 * c);
    }
    for (int i = tid; i < 32; i += blockDim.x) {
        float fr  = __expf(-((float)(2 * i) / 64.f) * LOG_BASE);
        float ang = (float)t * fr;
        float c = __cosf(ang), s = __sinf(ang);
        const unsigned short* kp = &kr[(size_t)bt * DROPE + 2 * i];
        float x1 = bf16_to_f32(kp[0]), x2 = bf16_to_f32(kp[1]);
        unsigned short o1 = f32_to_bf16(x1 * c - x2 * s);
        unsigned short o2 = f32_to_bf16(x1 * s + x2 * c);
        #pragma unroll
        for (int h = 0; h < NHEADS; ++h) {
            unsigned short* kb = &k[(size_t)bt * QROW + h * DQK + DHEAD + 2 * i];
            kb[0] = o1; kb[1] = o2;
        }
    }
}

// --------------------------- Flash attention -------------------------------
// grid = (SEQ/128, BATCH*NHEADS); block = 256 (8 waves). Each wave owns 16 q
// rows; online softmax over 64-wide causal kv tiles; S and P*V via WMMA.
__global__ void __launch_bounds__(256)
mla_flash_attn(const unsigned short* __restrict__ q,
               const unsigned short* __restrict__ k,
               const unsigned short* __restrict__ v,
               unsigned short* __restrict__ o)
{
    constexpr int QB = 128, KB = 64;
    __shared__ unsigned short Ks[KB][DQK + 8];   // K rows, 400B stride (16B ok)
    __shared__ unsigned short Vs[DHEAD][KB + 2]; // V transposed: [d][kv]
    __shared__ unsigned short Ps[QB][KB + 2];    // softmax probs (bf16)

    const int tid  = threadIdx.x;
    const int wave = tid >> 5, lane = tid & 31;
    const int bh = blockIdx.y;
    const int b  = bh >> 4, h = bh & 15;
    const int q0 = blockIdx.x * QB;

    const unsigned short* qbase = q + (size_t)b * SEQ * QROW + h * DQK;
    const unsigned short* kbase = k + (size_t)b * SEQ * QROW + h * DQK;
    const unsigned short* vbase = v + (size_t)b * SEQ * VROW + h * DHEAD;
    unsigned short*       obase = o + (size_t)b * SEQ * VROW + h * DHEAD;

    const int mfr  = lane & 15;
    const int hk   = (lane >> 4) * 8;
    const int bHlf = (lane >> 4) * 8;

    // Q fragments for this wave's 16 rows, K = 0..191 (6 fragments).
    AFrag qf[6];
    {
        const unsigned short* qr = qbase + (size_t)(q0 + wave * 16 + mfr) * QROW;
        #pragma unroll
        for (int ks = 0; ks < 6; ++ks) {
            int kb2 = ks * 32;
            #pragma unroll
            for (int vv = 0; vv < 4; ++vv)
                qf[ks].u[vv] = *(const unsigned int*)&qr[kb2 + hk + 2 * vv];
            #pragma unroll
            for (int vv = 0; vv < 4; ++vv)
                qf[ks].u[4 + vv] = *(const unsigned int*)&qr[kb2 + 16 + hk + 2 * vv];
        }
    }

    floatx8 zero = {};
    floatx8 oacc[8];
    #pragma unroll
    for (int dt = 0; dt < 8; ++dt) oacc[dt] = zero;
    float mrow[8], lrow[8];
    #pragma unroll
    for (int r = 0; r < 8; ++r) { mrow[r] = -3.0e38f; lrow[r] = 0.f; }

    const float scale = 0.072168783648703221f;   // 1/sqrt(192)
    const int rbase = q0 + wave * 16 + (lane >> 4) * 8;

    for (int kv0 = 0; kv0 < q0 + QB; kv0 += KB) {
        __syncthreads();   // previous tile fully consumed
        // ---- stage K tile (pure row-major copy -> async to LDS) ----
        {
            int r    = tid >> 2;
            int cseg = (tid & 3) * 48;
            const unsigned short* kp = &kbase[(size_t)(kv0 + r) * QROW + cseg];
#if USE_ASYNC_LDS
            #pragma unroll
            for (int c = 0; c < 6; ++c)
                async_copy16(kp + c * 8, &Ks[r][cseg + c * 8]);
#else
            unsigned int* ls = (unsigned int*)&Ks[r][cseg];
            const unsigned int* gp = (const unsigned int*)kp;
            #pragma unroll
            for (int c2 = 0; c2 < 24; ++c2) ls[c2] = gp[c2];
#endif
        }
        // ---- stage V tile transposed ----
        {
            int kvi  = tid >> 2;
            int dseg = (tid & 3) * 32;
            const uint4* vp =
                reinterpret_cast<const uint4*>(&vbase[(size_t)(kv0 + kvi) * VROW + dseg]);
            #pragma unroll
            for (int c = 0; c < 4; ++c) {
                uint4 d = vp[c];
                unsigned int w[4] = { d.x, d.y, d.z, d.w };
                #pragma unroll
                for (int e = 0; e < 4; ++e) {
                    int dcol = dseg + c * 8 + e * 2;
                    Vs[dcol][kvi]     = (unsigned short)(w[e] & 0xFFFFu);
                    Vs[dcol + 1][kvi] = (unsigned short)(w[e] >> 16);
                }
            }
        }
#if USE_ASYNC_LDS
        __builtin_amdgcn_s_wait_asynccnt(0);
#endif
        __syncthreads();

        // ---- S = Q * K^T  (4 N-tiles of 16) ----
        floatx8 s[4];
        #pragma unroll
        for (int j = 0; j < 4; ++j) {
            floatx8 sj = zero;
            #pragma unroll
            for (int ks = 0; ks < 6; ++ks) {
                AFrag bf;
                const unsigned short* kp = &Ks[j * 16 + (lane & 15)][ks * 32 + bHlf];
                #pragma unroll
                for (int vv = 0; vv < 8; ++vv)
                    bf.u[vv] = *(const unsigned int*)&kp[2 * vv + (vv >= 4 ? 8 : 0)];
                sj = __builtin_amdgcn_wmma_f32_16x16x32_bf16(
                    false, qf[ks].v, false, bf.v, (short)0, sj, false, false);
            }
            s[j] = sj;
        }

        // ---- scale + causal mask + online softmax ----
        #pragma unroll
        for (int r = 0; r < 8; ++r) {
            #pragma unroll
            for (int j = 0; j < 4; ++j) {
                int col = kv0 + j * 16 + (lane & 15);
                float val = s[j][r] * scale;
                if (col > rbase + r) val = -3.0e38f;
                s[j][r] = val;
            }
            float tm = fmaxf(fmaxf(s[0][r], s[1][r]), fmaxf(s[2][r], s[3][r]));
            #pragma unroll
            for (int msk = 8; msk >= 1; msk >>= 1)
                tm = fmaxf(tm, __shfl_xor(tm, msk, 32));
            float mnew  = fmaxf(mrow[r], tm);
            float alpha = __expf(mrow[r] - mnew);
            mrow[r] = mnew;
            float rsum = 0.f;
            #pragma unroll
            for (int j = 0; j < 4; ++j) {
                float p = __expf(s[j][r] - mnew);
                s[j][r] = p;
                rsum += p;
            }
            #pragma unroll
            for (int msk = 8; msk >= 1; msk >>= 1)
                rsum += __shfl_xor(rsum, msk, 32);
            lrow[r] = lrow[r] * alpha + rsum;
            #pragma unroll
            for (int dt = 0; dt < 8; ++dt)
                oacc[dt][r] = oacc[dt][r] * alpha;
        }

        // ---- P -> LDS (bf16) ----
        {
            int prow = wave * 16 + (lane >> 4) * 8;
            #pragma unroll
            for (int j = 0; j < 4; ++j)
                #pragma unroll
                for (int r = 0; r < 8; ++r)
                    Ps[prow + r][j * 16 + (lane & 15)] = f32_to_bf16(s[j][r]);
        }
        __syncthreads();

        // ---- O += P * V ----
        AFrag pf[2];
        #pragma unroll
        for (int ks = 0; ks < 2; ++ks) {
            const unsigned short* pr = &Ps[wave * 16 + mfr][ks * 32];
            #pragma unroll
            for (int vv = 0; vv < 4; ++vv)
                pf[ks].u[vv] = *(const unsigned int*)&pr[hk + 2 * vv];
            #pragma unroll
            for (int vv = 0; vv < 4; ++vv)
                pf[ks].u[4 + vv] = *(const unsigned int*)&pr[16 + hk + 2 * vv];
        }
        #pragma unroll
        for (int dt = 0; dt < 8; ++dt)
            #pragma unroll
            for (int ks = 0; ks < 2; ++ks) {
                AFrag vf;
                const unsigned short* vp = &Vs[dt * 16 + (lane & 15)][ks * 32 + bHlf];
                #pragma unroll
                for (int vv = 0; vv < 8; ++vv)
                    vf.u[vv] = *(const unsigned int*)&vp[2 * vv + (vv >= 4 ? 8 : 0)];
                oacc[dt] = __builtin_amdgcn_wmma_f32_16x16x32_bf16(
                    false, pf[ks].v, false, vf.v, (short)0, oacc[dt], false, false);
            }
    }

    // ---- normalize + store (bf16) ----
    #pragma unroll
    for (int r = 0; r < 8; ++r) {
        float inv = 1.0f / lrow[r];
        int row = rbase + r;
        #pragma unroll
        for (int dt = 0; dt < 8; ++dt) {
            int d = dt * 16 + (lane & 15);
            obase[(size_t)row * VROW + d] = f32_to_bf16(oacc[dt][r] * inv);
        }
    }
}

// ------------------------------ launcher -----------------------------------
extern "C" void kernel_launch(void* const* d_in, const int* in_sizes, int n_in,
                              void* d_out, int out_size, void* d_ws, size_t ws_size,
                              hipStream_t stream)
{
    (void)in_sizes; (void)n_in; (void)out_size; (void)ws_size;
    const float* hidden = (const float*)d_in[0];
    // d_in[1] = attention_mask (deterministic causal; applied analytically)
    const float* W_dq  = (const float*)d_in[2];
    const float* W_uq  = (const float*)d_in[3];
    const float* W_dkv = (const float*)d_in[4];
    const float* W_uk  = (const float*)d_in[5];
    const float* W_uv  = (const float*)d_in[6];
    const float* W_qr  = (const float*)d_in[7];
    const float* W_kr  = (const float*)d_in[8];
    const float* W_o   = (const float*)d_in[9];
    float* out = (float*)d_out;

    unsigned short* ws = (unsigned short*)d_ws;
    size_t off = 0;
    auto alloc = [&](size_t n) { unsigned short* p = ws + off; off += n; return p; };

    unsigned short* hbf   = alloc((size_t)MTOT * DMODEL);
    unsigned short* wdq   = alloc((size_t)DMODEL * DLATQ);
    unsigned short* wuq   = alloc((size_t)DLATQ * DMODEL);
    unsigned short* wqr   = alloc((size_t)DLATQ * NHEADS * DROPE);
    unsigned short* wdkv  = alloc((size_t)DMODEL * DLAT);
    unsigned short* wuk   = alloc((size_t)DLAT * DMODEL);
    unsigned short* wuv   = alloc((size_t)DLAT * DMODEL);
    unsigned short* wkr   = alloc((size_t)DMODEL * DROPE);
    unsigned short* wo    = alloc((size_t)DMODEL * DMODEL);
    unsigned short* cq    = alloc((size_t)MTOT * DLATQ);
    unsigned short* ckv   = alloc((size_t)MTOT * DLAT);
    unsigned short* qbuf  = alloc((size_t)MTOT * QROW);
    unsigned short* kbuf  = alloc((size_t)MTOT * QROW);
    unsigned short* krtmp = alloc((size_t)MTOT * DROPE);
    unsigned short* vbuf  = alloc((size_t)MTOT * VROW);
    unsigned short* aobuf = alloc((size_t)MTOT * VROW);

    auto cvt = [&](const float* s, unsigned short* d, size_t n) {
        cvt_f32_bf16_kernel<<<dim3(2048), dim3(256), 0, stream>>>(s, d, (int)n);
    };
    cvt(hidden, hbf, (size_t)MTOT * DMODEL);
    cvt(W_dq,  wdq,  (size_t)DMODEL * DLATQ);
    cvt(W_uq,  wuq,  (size_t)DLATQ * DMODEL);
    cvt(W_qr,  wqr,  (size_t)DLATQ * NHEADS * DROPE);
    cvt(W_dkv, wdkv, (size_t)DMODEL * DLAT);
    cvt(W_uk,  wuk,  (size_t)DLAT * DMODEL);
    cvt(W_uv,  wuv,  (size_t)DLAT * DMODEL);
    cvt(W_kr,  wkr,  (size_t)DMODEL * DROPE);
    cvt(W_o,   wo,   (size_t)DMODEL * DMODEL);

    dim3 blk(256);
    auto grid = [](int M, int N) { return dim3(M / 128, (N + 127) / 128); };

    // c_q = hidden @ W_dq
    gemm_bf16_wmma<false><<<grid(MTOT, DLATQ), blk, 0, stream>>>(
        hbf, wdq, cq, MTOT, DLATQ, DMODEL, DLATQ, DLATQ, 0, 0);
    // q_c -> qbuf[..., h, 0:128]
    gemm_bf16_wmma<false><<<grid(MTOT, DMODEL), blk, 0, stream>>>(
        cq, wuq, qbuf, MTOT, DMODEL, DLATQ, QROW, DHEAD, DQK, 0);
    // q_r (pre-rope) -> qbuf[..., h, 128:192]
    gemm_bf16_wmma<false><<<grid(MTOT, NHEADS * DROPE), blk, 0, stream>>>(
        cq, wqr, qbuf, MTOT, NHEADS * DROPE, DLATQ, QROW, DROPE, DQK, DHEAD);
    // c_kv = hidden @ W_dkv
    gemm_bf16_wmma<false><<<grid(MTOT, DLAT), blk, 0, stream>>>(
        hbf, wdkv, ckv, MTOT, DLAT, DMODEL, DLAT, DLAT, 0, 0);
    // k_c -> kbuf[..., h, 0:128]
    gemm_bf16_wmma<false><<<grid(MTOT, DMODEL), blk, 0, stream>>>(
        ckv, wuk, kbuf, MTOT, DMODEL, DLAT, QROW, DHEAD, DQK, 0);
    // v -> vbuf[..., h, 0:128]
    gemm_bf16_wmma<false><<<grid(MTOT, DMODEL), blk, 0, stream>>>(
        ckv, wuv, vbuf, MTOT, DMODEL, DLAT, VROW, DMODEL, 0, 0);
    // k_r (pre-rope, shared across heads)
    gemm_bf16_wmma<false><<<grid(MTOT, DROPE), blk, 0, stream>>>(
        hbf, wkr, krtmp, MTOT, DROPE, DMODEL, DROPE, DROPE, 0, 0);

    rope_kernel<<<dim3(MTOT), blk, 0, stream>>>(qbuf, kbuf, krtmp);

    mla_flash_attn<<<dim3(SEQ / 128, BATCH * NHEADS), blk, 0, stream>>>(
        qbuf, kbuf, vbuf, aobuf);

    // out = attn @ W_o   (fp32 output)
    gemm_bf16_wmma<true><<<grid(MTOT, DMODEL), blk, 0, stream>>>(
        aobuf, wo, out, MTOT, DMODEL, DMODEL, DMODEL, DMODEL, 0, 0);
}
